// SeqKNNAttn_32899449487852
// MI455X (gfx1250) — compile-verified, matrix operand order checked by reading, not run on hardware
//
#include <hip/hip_runtime.h>

// ---------------------------------------------------------------------------
// SeqKNNAttn on MI455X (gfx1250, wave32, WMMA)
//
//   idx(n)  = contiguous window lo = clamp(n-8, 0, N-16) .. lo+15
//             (top_k ties pick smaller index -> n-8 over n+8; softmax over the
//              group axis is permutation invariant so only the set matters)
//   qkv     = x @ w_qkv^T   (16384x512 x 512x1536)  -> v_wmma bf16, LDS-staged W
//   attn    = softmax(q.k_win/8); out = attn @ v_win -> VALU (1.5% of FLOPs)
//   out     = attn_out @ w_proj^T + b                -> v_wmma bf16, LDS-staged W
// ---------------------------------------------------------------------------

typedef unsigned short u16;
typedef unsigned int   u32;

typedef __attribute__((ext_vector_type(16))) __bf16 v16bf;
typedef __attribute__((ext_vector_type(8)))  __bf16 v8bf;
typedef __attribute__((ext_vector_type(8)))  float  v8f;

#define B_    4
#define N_    4096
#define D_    512
#define H_    8
#define DH_   64
#define G_    16
#define ROWS  (B_ * N_)     // 16384
#define E3    (3 * D_)      // 1536

// ------------------------------- helpers -----------------------------------

__device__ __forceinline__ u16 f2bf(float f) {          // round-to-nearest-even
    u32 u = __float_as_uint(f);
    u32 r = u + 0x7FFFu + ((u >> 16) & 1u);
    return (u16)(r >> 16);
}
__device__ __forceinline__ float bflo(u32 u) { return __uint_as_float(u << 16); }
__device__ __forceinline__ float bfhi(u32 u) { return __uint_as_float(u & 0xFFFF0000u); }

// 16-element bf16 WMMA fragment = two contiguous 16B chunks at p and p+16.
// ISA 16-bit 16x32 layout: lane holds K = kbase+0..7 and kbase+16..23 where
// kbase = (lane>=16 ? 8 : 0); caller passes p = row_base + k0 + kbase.
__device__ __forceinline__ v16bf frag2(const u16* p) {
    v8bf lo = *reinterpret_cast<const v8bf*>(p);
    v8bf hi = *reinterpret_cast<const v8bf*>(p + 16);
    return __builtin_shufflevector(lo, hi, 0,1,2,3,4,5,6,7,8,9,10,11,12,13,14,15);
}

// --------------------------- f32 -> bf16 cast ------------------------------

__global__ void cast_bf16_kernel(const float* __restrict__ src,
                                 u16* __restrict__ dst, int n) {
    int i      = blockIdx.x * blockDim.x + threadIdx.x;
    int stride = gridDim.x * blockDim.x;
    for (; i < n; i += stride) dst[i] = f2bf(src[i]);
}

// ------------------------------ WMMA GEMM ----------------------------------
// C[ROWS x NTOT] = A[ROWS x 512](bf16) * W[NTOT x 512]^T(bf16)  (+bias)
// Block (256 thr = 8 waves): 256(M) x 64(N) tile; the 64x512 W tile is staged
// in LDS once and shared by all waves (ds_load_b128 B-fragments).
// Wave: 32(M) x 64(N) -> 2 A-frags x 4 B-frags -> 8 independent WMMAs / K-step.

template <int NTOT, bool OUT_BF16, bool ADD_BIAS>
__global__ void wmma_gemm_kernel(const u16* __restrict__ A,
                                 const u16* __restrict__ W,
                                 const float* __restrict__ bias,
                                 void* __restrict__ Cout) {
    __shared__ u16 wtile[64 * D_];                      // 64 KB

    const int nblk   = NTOT / 64;
    const int rowblk = blockIdx.x / nblk;
    const int ng     = blockIdx.x % nblk;

    // cooperative 64x512 bf16 weight-tile load: 4096 x 16B, 16 chunks/thread
    {
        const uint4* gsrc = reinterpret_cast<const uint4*>(W + (size_t)ng * 64 * D_);
        uint4* ldst       = reinterpret_cast<uint4*>(wtile);
#pragma unroll
        for (int i = 0; i < 16; ++i) {
            const int c = threadIdx.x + 256 * i;
            ldst[c] = gsrc[c];
        }
    }
    __syncthreads();

    const int lane  = threadIdx.x & 31;
    const int wv    = threadIdx.x >> 5;                 // 0..7
    const int ln    = lane & 15;
    const int khalf = (lane >> 4) * 8;

    const int m0    = rowblk * 256 + wv * 32;           // wave's first row
    const u16* arow0 = A + (size_t)(m0 + ln) * D_;
    const u16* arow1 = A + (size_t)(m0 + 16 + ln) * D_;

    v8f acc0[4] = {};
    v8f acc1[4] = {};

    for (int k0 = 0; k0 < D_; k0 += 32) {
        const int ka = k0 + khalf;
        v16bf a0 = frag2(arow0 + ka);
        v16bf a1 = frag2(arow1 + ka);
        __builtin_prefetch(arow0 + ka + 32, 0, 1);      // global_prefetch_b8
        __builtin_prefetch(arow1 + ka + 32, 0, 1);
#pragma unroll
        for (int j = 0; j < 4; ++j) {
            v16bf b = frag2(&wtile[(j * 16 + ln) * D_ + ka]);   // ds_load_b128 x2
            acc0[j] = __builtin_amdgcn_wmma_f32_16x16x32_bf16(
                false, a0, false, b, (short)0, acc0[j], false, false);
            acc1[j] = __builtin_amdgcn_wmma_f32_16x16x32_bf16(
                false, a1, false, b, (short)0, acc1[j], false, false);
        }
    }

    // C/D layout: lanes 0-15 -> M = vgpr(0..7), lanes 16-31 -> M = 8+vgpr; N = lane%16
    const int msub = (lane >> 4) * 8;
#pragma unroll
    for (int j = 0; j < 4; ++j) {
        const int col = ng * 64 + j * 16 + ln;
        const float bb = ADD_BIAS ? bias[col] : 0.0f;
#pragma unroll
        for (int r = 0; r < 8; ++r) {
            const float v0 = acc0[j][r] + bb;
            const float v1 = acc1[j][r] + bb;
            const size_t o0 = (size_t)(m0 + msub + r) * NTOT + col;
            const size_t o1 = (size_t)(m0 + 16 + msub + r) * NTOT + col;
            if (OUT_BF16) { ((u16*)Cout)[o0] = f2bf(v0); ((u16*)Cout)[o1] = f2bf(v1); }
            else          { ((float*)Cout)[o0] = v0;     ((float*)Cout)[o1] = v1;     }
        }
    }
}

// -------------------------- windowed attention -----------------------------
// One thread per (b, n, h). Window = [clamp(n-8,0,N-16) .. +15].

__global__ void window_attn_kernel(const u16* __restrict__ qkv,
                                   u16* __restrict__ outa) {
    const int t = blockIdx.x * blockDim.x + threadIdx.x;
    const int h = t & (H_ - 1);
    const int n = (t >> 3) & (N_ - 1);
    const int b = t >> 15;               // / (H_ * N_)

    int lo = n - 8;
    lo = lo < 0 ? 0 : lo;
    lo = lo > (N_ - G_) ? (N_ - G_) : lo;

    const u32* qp = reinterpret_cast<const u32*>(
        qkv + (size_t)(b * N_ + n) * E3 + h * DH_);

    u32 q[32];
#pragma unroll
    for (int c = 0; c < 32; ++c) q[c] = qp[c];

    float lg[G_];
    float mx = -1e30f;
#pragma unroll 4
    for (int g = 0; g < G_; ++g) {
        const u32* kr = reinterpret_cast<const u32*>(
            qkv + (size_t)(b * N_ + lo + g) * E3 + D_ + h * DH_);
        float s = 0.0f;
#pragma unroll
        for (int c = 0; c < 32; ++c) {
            const u32 kv = kr[c];
            s = fmaf(bflo(q[c]), bflo(kv), s);
            s = fmaf(bfhi(q[c]), bfhi(kv), s);
        }
        s *= 0.125f;                     // 1/sqrt(64)
        lg[g] = s;
        mx = fmaxf(mx, s);
    }

    float sum = 0.0f;
#pragma unroll
    for (int g = 0; g < G_; ++g) { lg[g] = __expf(lg[g] - mx); sum += lg[g]; }
    const float inv = 1.0f / sum;

    float o[DH_];
#pragma unroll
    for (int i = 0; i < DH_; ++i) o[i] = 0.0f;

#pragma unroll 4
    for (int g = 0; g < G_; ++g) {
        const float wgt = lg[g] * inv;
        const u32* vr = reinterpret_cast<const u32*>(
            qkv + (size_t)(b * N_ + lo + g) * E3 + 2 * D_ + h * DH_);
#pragma unroll
        for (int c = 0; c < 32; ++c) {
            const u32 vv = vr[c];
            o[2 * c]     = fmaf(wgt, bflo(vv), o[2 * c]);
            o[2 * c + 1] = fmaf(wgt, bfhi(vv), o[2 * c + 1]);
        }
    }

    u32* op = reinterpret_cast<u32*>(outa + (size_t)(b * N_ + n) * D_ + h * DH_);
#pragma unroll
    for (int c = 0; c < 32; ++c)
        op[c] = ((u32)f2bf(o[2 * c + 1]) << 16) | (u32)f2bf(o[2 * c]);
}

// ------------------------------- launcher ----------------------------------

extern "C" void kernel_launch(void* const* d_in, const int* in_sizes, int n_in,
                              void* d_out, int out_size, void* d_ws, size_t ws_size,
                              hipStream_t stream) {
    const float* x      = (const float*)d_in[0];
    // d_in[1] = z : unused by the reference
    const float* w_qkv  = (const float*)d_in[2];
    const float* w_proj = (const float*)d_in[3];
    const float* b_proj = (const float*)d_in[4];
    float* out = (float*)d_out;

    char* ws = (char*)d_ws;
    u16* x_bf     = (u16*)(ws + 0);                  // 16 MB   [ROWS x 512]
    u16* wqkv_bf  = (u16*)(ws + 16777216);           // 1.5 MB  [1536 x 512]
    u16* wproj_bf = (u16*)(ws + 18350080);           // 0.5 MB  [512 x 512]
    u16* qkv_bf   = (u16*)(ws + 18874368);           // 48 MB   [ROWS x 1536]
    u16* attn_bf  = x_bf;                            // reuse: x_bf dead post-GEMM1

    // 1) casts to bf16
    cast_bf16_kernel<<<2048, 256, 0, stream>>>(x,      x_bf,     ROWS * D_);
    cast_bf16_kernel<<<768,  256, 0, stream>>>(w_qkv,  wqkv_bf,  E3 * D_);
    cast_bf16_kernel<<<256,  256, 0, stream>>>(w_proj, wproj_bf, D_ * D_);

    // 2) QKV projection: blocks = (16384/256) * (1536/64) = 64*24 = 1536
    wmma_gemm_kernel<E3, true, false>
        <<<(ROWS / 256) * (E3 / 64), 256, 0, stream>>>(x_bf, wqkv_bf, nullptr, qkv_bf);

    // 3) windowed attention: one thread per (b,n,h)
    window_attn_kernel<<<(ROWS * H_) / 256, 256, 0, stream>>>(qkv_bf, attn_bf);

    // 4) output projection + bias: blocks = 64*8 = 512
    wmma_gemm_kernel<D_, false, true>
        <<<(ROWS / 256) * (D_ / 64), 256, 0, stream>>>(attn_bf, wproj_bf, b_proj, out);
}
